// DeepseekV2MoE_70875550319379
// MI455X (gfx1250) — compile-verified, matrix operand order checked by reading, not run on hardware
//
#include <hip/hip_runtime.h>
#include <hip/hip_bf16.h>
#include <math.h>

// ---- problem constants (from reference) ----
#define EXPERTS 8
#define TOPK    2
#define HID     1024
#define IMID    512
#define SIMID   512

typedef __attribute__((ext_vector_type(16))) __bf16         v16bf;
typedef __attribute__((ext_vector_type(8)))  float          v8f;
typedef __attribute__((ext_vector_type(8)))  unsigned int   v8u;
typedef __attribute__((ext_vector_type(8)))  unsigned short us8;
typedef __attribute__((ext_vector_type(2)))  unsigned int   u32x2;

// fp32 -> bf16 with round-to-nearest-even
__device__ __forceinline__ unsigned short f2bf(float f) {
    unsigned int b = __builtin_bit_cast(unsigned int, f);
    unsigned int r = b + 0x7FFFu + ((b >> 16) & 1u);
    return (unsigned short)(r >> 16);
}

__device__ __forceinline__ v8f wmma_bf16(v16bf a, v16bf b, v8f c) {
    return __builtin_amdgcn_wmma_f32_16x16x32_bf16(false, a, false, b, (short)0, c,
                                                   false, false);
}

// Load a 16x32 bf16 A/B fragment from an LDS tile stored [row][k] (stride 40).
// ISA layout: VGPR j holds K = (j<4 ? 2j : 16+2(j-4)) + 8*half, pair (K, K+1);
// row (M for A, N for B) = lane & 15, half = lane >> 4.
__device__ __forceinline__ v16bf load_frag(const unsigned short S[][40], int row, int half) {
    v8u u;
#pragma unroll
    for (int j = 0; j < 8; ++j) {
        int kb = ((j < 4) ? (2 * j) : (16 + 2 * (j - 4))) + half * 8;
        unsigned int lo = S[row][kb];
        unsigned int hi = S[row][kb + 1];
        u[j] = lo | (hi << 16);
    }
    return __builtin_bit_cast(v16bf, u);
}

__device__ __forceinline__ float gelu_exact(float g) {
    return 0.5f * g * (1.0f + erff(g * 0.70710678118654752f));
}

// ------------------------------------------------------------------
// fp32 -> bf16 elementwise (vec4 in, 8 bytes out per thread)
// ------------------------------------------------------------------
__global__ __launch_bounds__(256) void cvt_kernel(const float* __restrict__ in,
                                                  unsigned short* __restrict__ out,
                                                  size_t n4) {
    size_t i = (size_t)blockIdx.x * blockDim.x + threadIdx.x;
    if (i >= n4) return;
    float4 v = ((const float4*)in)[i];
    u32x2 u;
    u.x = (unsigned int)f2bf(v.x) | ((unsigned int)f2bf(v.y) << 16);
    u.y = (unsigned int)f2bf(v.z) | ((unsigned int)f2bf(v.w) << 16);
    ((u32x2*)out)[i] = u;
}

// ------------------------------------------------------------------
// fp32 [b][R][C] row-major -> bf16 [b][C][R] row-major (transpose+convert)
// 32x32 LDS tiles; R, C multiples of 32.
// ------------------------------------------------------------------
__global__ __launch_bounds__(256) void transpose_cvt_kernel(
    const float* __restrict__ in, unsigned short* __restrict__ out, int R, int C) {
    __shared__ unsigned short tile[32][33];
    const float* src = in + (size_t)blockIdx.z * R * C;
    unsigned short* dst = out + (size_t)blockIdx.z * R * C;
    int r0 = blockIdx.y * 32, c0 = blockIdx.x * 32;
    int tr = threadIdx.x >> 5;  // 0..7
    int tc = threadIdx.x & 31;  // 0..31
#pragma unroll
    for (int i = 0; i < 4; ++i) {
        int r = tr + i * 8;
        tile[r][tc] = f2bf(src[(size_t)(r0 + r) * C + (c0 + tc)]);
    }
    __syncthreads();
#pragma unroll
    for (int i = 0; i < 4; ++i) {
        int c = tr + i * 8;
        dst[(size_t)(c0 + c) * R + (r0 + tc)] = tile[tc][c];
    }
}

// ------------------------------------------------------------------
// zero the expert counters (ws is poisoned by the harness)
// ------------------------------------------------------------------
__global__ void init_counts_kernel(int* counts) {
    if (threadIdx.x < EXPERTS) counts[threadIdx.x] = 0;
}

// ------------------------------------------------------------------
// router: logits -> softmax -> greedy top-2 -> renormalized weights
// one wave32 per token
// ------------------------------------------------------------------
__global__ __launch_bounds__(256) void router_kernel(
    const float* __restrict__ x, const float* __restrict__ gw, int T,
    int* __restrict__ topk_idx, float* __restrict__ topk_w, int* __restrict__ counts) {
    int wave = threadIdx.x >> 5, lane = threadIdx.x & 31;
    int t = blockIdx.x * 8 + wave;
    if (t >= T) return;
    float acc[EXPERTS];
#pragma unroll
    for (int e = 0; e < EXPERTS; ++e) acc[e] = 0.f;
    const float* xr = x + (size_t)t * HID;
    for (int h = lane; h < HID; h += 32) {
        float xv = xr[h];
#pragma unroll
        for (int e = 0; e < EXPERTS; ++e) acc[e] += xv * gw[e * HID + h];
    }
#pragma unroll
    for (int off = 16; off > 0; off >>= 1)
#pragma unroll
        for (int e = 0; e < EXPERTS; ++e) acc[e] += __shfl_xor(acc[e], off, 32);
    if (lane == 0) {
        float mx = acc[0];
#pragma unroll
        for (int e = 1; e < EXPERTS; ++e) mx = fmaxf(mx, acc[e]);
        float p[EXPERTS], s = 0.f;
#pragma unroll
        for (int e = 0; e < EXPERTS; ++e) { p[e] = expf(acc[e] - mx); s += p[e]; }
        float inv = 1.f / s;
#pragma unroll
        for (int e = 0; e < EXPERTS; ++e) p[e] *= inv;
        int e1 = 0;
#pragma unroll
        for (int e = 1; e < EXPERTS; ++e) if (p[e] > p[e1]) e1 = e;
        int e2 = (e1 == 0) ? 1 : 0;
#pragma unroll
        for (int e = 0; e < EXPERTS; ++e) if (e != e1 && p[e] > p[e2]) e2 = e;
        float w1 = p[e1], w2 = p[e2];
        float d = 1.f / (w1 + w2 + 1e-20f);
        topk_idx[2 * t + 0] = e1; topk_idx[2 * t + 1] = e2;
        topk_w[2 * t + 0] = w1 * d; topk_w[2 * t + 1] = w2 * d;
        atomicAdd(&counts[e1], 1);
        atomicAdd(&counts[e2], 1);
    }
}

// ------------------------------------------------------------------
// exclusive prefix scan over 8 counts
// ------------------------------------------------------------------
__global__ void scan_kernel(const int* counts, int* offsets, int* cursors) {
    if (threadIdx.x == 0 && blockIdx.x == 0) {
        int acc = 0;
        for (int e = 0; e < EXPERTS; ++e) {
            offsets[e] = acc;
            cursors[e] = acc;
            acc += counts[e];
        }
        offsets[EXPERTS] = acc;
    }
}

// ------------------------------------------------------------------
// scatter (token, weight) pairs into compact per-expert lists
// ------------------------------------------------------------------
__global__ __launch_bounds__(256) void scatter_kernel(
    const int* __restrict__ topk_idx, const float* __restrict__ topk_w, int T,
    int* __restrict__ cursors, int* __restrict__ pair_tok, float* __restrict__ pair_w) {
    int t = blockIdx.x * blockDim.x + threadIdx.x;
    if (t >= T) return;
#pragma unroll
    for (int k = 0; k < TOPK; ++k) {
        int e = topk_idx[2 * t + k];
        int p = atomicAdd(&cursors[e], 1);
        pair_tok[p] = t;
        pair_w[p] = topk_w[2 * t + k];
    }
}

// ------------------------------------------------------------------
// fused gate+up projection + exact GELU*up, bf16 WMMA.
// All operands pre-converted bf16; weights pre-transposed to [N][K].
// Out-of-range M rows are CLAMPED (not zeroed): their results are never
// stored, and WMMA does not mix rows across M, so garbage is harmless.
// GATHER=true : per-expert gathered token rows (blockIdx.z = expert)
// GATHER=false: shared expert over all T tokens
// ------------------------------------------------------------------
template <bool GATHER>
__global__ __launch_bounds__(256) void gateup_kernel(
    const unsigned short* __restrict__ xbf,   // [T][HID] bf16
    const unsigned short* __restrict__ WgT,   // [e][IMID][HID] bf16
    const unsigned short* __restrict__ WuT,   // [e][IMID][HID] bf16
    const int* __restrict__ pair_tok, const int* __restrict__ offsets,
    const int* __restrict__ counts, int fixed_count,
    unsigned short* __restrict__ inter) {
    const int e = GATHER ? blockIdx.z : 0;
    const int cnt = GATHER ? counts[e] : fixed_count;
    const int m0 = blockIdx.x * 64;
    if (m0 >= cnt) return;
    const int n0 = blockIdx.y * 64;
    const int rowbase = GATHER ? offsets[e] : 0;
    const unsigned short* Wg = WgT + (GATHER ? (size_t)e * HID * IMID : 0);
    const unsigned short* Wu = WuT + (GATHER ? (size_t)e * HID * IMID : 0);

    __shared__ unsigned short As[64][40];
    __shared__ unsigned short Bg[64][40];
    __shared__ unsigned short Bu[64][40];

    const int tid = threadIdx.x;
    const int lane = tid & 31, wave = tid >> 5;
    const int mi = wave & 3, nb = (wave >> 2) * 2;
    const int lm = lane & 15, half = lane >> 4;

    // Per-thread tile-fill assignment: row = tid>>2 (0..63), k-group = (tid&3)*8.
    const int trow = tid >> 2;
    const int tkg = (tid & 3) * 8;
    const int arow = min(m0 + trow, cnt - 1);  // clamp: OOR rows never stored
    const int tok = GATHER ? pair_tok[rowbase + arow] : arow;
    const size_t abase = (size_t)tok * HID + tkg;
    const size_t bbase = (size_t)(n0 + trow) * HID + tkg;

    v8f accg0 = {}, accg1 = {}, accu0 = {}, accu1 = {};

    for (int k0 = 0; k0 < HID; k0 += 32) {
        us8 av = *(const us8*)&xbf[abase + k0];
        us8 gv = *(const us8*)&Wg[bbase + k0];
        us8 uv = *(const us8*)&Wu[bbase + k0];
        if (k0 + 32 < HID) {  // prefetch next k-tile of the weight streams
            __builtin_prefetch(&Wg[bbase + k0 + 32], 0, 1);
            __builtin_prefetch(&Wu[bbase + k0 + 32], 0, 1);
        }
        *(us8*)&As[trow][tkg] = av;
        *(us8*)&Bg[trow][tkg] = gv;
        *(us8*)&Bu[trow][tkg] = uv;
        __syncthreads();
        v16bf a  = load_frag(As, mi * 16 + lm, half);
        v16bf g0 = load_frag(Bg, (nb + 0) * 16 + lm, half);
        v16bf g1 = load_frag(Bg, (nb + 1) * 16 + lm, half);
        v16bf u0 = load_frag(Bu, (nb + 0) * 16 + lm, half);
        v16bf u1 = load_frag(Bu, (nb + 1) * 16 + lm, half);
        accg0 = wmma_bf16(a, g0, accg0);
        accg1 = wmma_bf16(a, g1, accg1);
        accu0 = wmma_bf16(a, u0, accu0);
        accu1 = wmma_bf16(a, u1, accu1);
        __syncthreads();
    }
    // epilogue: inter = gelu(g) * u, stored bf16 (compact pair rows)
#pragma unroll
    for (int r = 0; r < 8; ++r) {
        int m = mi * 16 + r + half * 8;
        int row = m0 + m;
        if (row < cnt) {
            size_t orow = (size_t)(rowbase + row) * IMID;
            float v0 = gelu_exact(accg0[r]) * accu0[r];
            float v1 = gelu_exact(accg1[r]) * accu1[r];
            inter[orow + n0 + (nb + 0) * 16 + lm] = f2bf(v0);
            inter[orow + n0 + (nb + 1) * 16 + lm] = f2bf(v1);
        }
    }
}

// ------------------------------------------------------------------
// down projection, bf16 WMMA. Weight pre-transposed to [N=H][K=I].
// SCATTER=true : scale by routing weight, atomicAdd into d_out
// SCATTER=false: shared expert, plain store (writes all of d_out first)
// ------------------------------------------------------------------
template <bool SCATTER>
__global__ __launch_bounds__(256) void down_kernel(
    const unsigned short* __restrict__ inter,  // [rows][IMID] bf16
    const unsigned short* __restrict__ WdT,    // [e][HID][IMID] bf16
    const int* __restrict__ pair_tok, const float* __restrict__ pair_w,
    const int* __restrict__ offsets, const int* __restrict__ counts, int fixed_count,
    float* __restrict__ out) {
    const int e = SCATTER ? blockIdx.z : 0;
    const int cnt = SCATTER ? counts[e] : fixed_count;
    const int m0 = blockIdx.x * 64;
    if (m0 >= cnt) return;
    const int n0 = blockIdx.y * 64;
    const int rowbase = SCATTER ? offsets[e] : 0;
    const unsigned short* Wd = WdT + (SCATTER ? (size_t)e * IMID * HID : 0);

    __shared__ unsigned short As[64][40];
    __shared__ unsigned short Bs[64][40];

    const int tid = threadIdx.x;
    const int lane = tid & 31, wave = tid >> 5;
    const int mi = wave & 3, nb = (wave >> 2) * 2;
    const int lm = lane & 15, half = lane >> 4;

    const int trow = tid >> 2;
    const int tkg = (tid & 3) * 8;
    const int arow = min(m0 + trow, cnt - 1);  // clamp: OOR rows never stored
    const size_t abase = (size_t)(rowbase + arow) * IMID + tkg;
    const size_t bbase = (size_t)(n0 + trow) * IMID + tkg;

    v8f acc0 = {}, acc1 = {};

    for (int k0 = 0; k0 < IMID; k0 += 32) {
        us8 av = *(const us8*)&inter[abase + k0];
        us8 bv = *(const us8*)&Wd[bbase + k0];
        if (k0 + 32 < IMID) {
            __builtin_prefetch(&Wd[bbase + k0 + 32], 0, 1);
        }
        *(us8*)&As[trow][tkg] = av;
        *(us8*)&Bs[trow][tkg] = bv;
        __syncthreads();
        v16bf a  = load_frag(As, mi * 16 + lm, half);
        v16bf b0 = load_frag(Bs, (nb + 0) * 16 + lm, half);
        v16bf b1 = load_frag(Bs, (nb + 1) * 16 + lm, half);
        acc0 = wmma_bf16(a, b0, acc0);
        acc1 = wmma_bf16(a, b1, acc1);
        __syncthreads();
    }
#pragma unroll
    for (int r = 0; r < 8; ++r) {
        int m = mi * 16 + r + half * 8;
        int row = m0 + m;
        if (row < cnt) {
            int col0 = n0 + (nb + 0) * 16 + lm;
            int col1 = n0 + (nb + 1) * 16 + lm;
            if (SCATTER) {
                int prow = rowbase + row;
                int tok = pair_tok[prow];
                float w = pair_w[prow];
                atomicAdd(&out[(size_t)tok * HID + col0], acc0[r] * w);
                atomicAdd(&out[(size_t)tok * HID + col1], acc1[r] * w);
            } else {
                out[(size_t)row * HID + col0] = acc0[r];
                out[(size_t)row * HID + col1] = acc1[r];
            }
        }
    }
}

// ------------------------------------------------------------------
extern "C" void kernel_launch(void* const* d_in, const int* in_sizes, int n_in,
                              void* d_out, int out_size, void* d_ws, size_t ws_size,
                              hipStream_t stream) {
    (void)n_in; (void)out_size; (void)ws_size;
    const float* x      = (const float*)d_in[0];
    const float* gate_w = (const float*)d_in[1];
    const float* Wg     = (const float*)d_in[2];
    const float* Wu     = (const float*)d_in[3];
    const float* Wd     = (const float*)d_in[4];
    const float* sWg    = (const float*)d_in[5];
    const float* sWu    = (const float*)d_in[6];
    const float* sWd    = (const float*)d_in[7];
    float* out = (float*)d_out;

    const int T = in_sizes[0] / HID;  // B*S tokens

    // workspace carve-out (256B aligned regions)
    char* ws = (char*)d_ws;
    size_t off = 0;
    auto carve = [&](size_t bytes) -> char* {
        char* p = ws + off;
        off += (bytes + 255) & ~(size_t)255;
        return p;
    };
    int*            counts   = (int*)carve(EXPERTS * 4);
    int*            offsets  = (int*)carve((EXPERTS + 1) * 4);
    int*            cursors  = (int*)carve(EXPERTS * 4);
    int*            topk_idx = (int*)carve((size_t)T * 2 * 4);
    float*          topk_w   = (float*)carve((size_t)T * 2 * 4);
    int*            pair_tok = (int*)carve((size_t)T * 2 * 4);
    float*          pair_w   = (float*)carve((size_t)T * 2 * 4);
    unsigned short* x_bf     = (unsigned short*)carve((size_t)T * HID * 2);
    unsigned short* WgT      = (unsigned short*)carve((size_t)EXPERTS * HID * IMID * 2);
    unsigned short* WuT      = (unsigned short*)carve((size_t)EXPERTS * HID * IMID * 2);
    unsigned short* WdT      = (unsigned short*)carve((size_t)EXPERTS * IMID * HID * 2);
    unsigned short* sWgT     = (unsigned short*)carve((size_t)HID * SIMID * 2);
    unsigned short* sWuT     = (unsigned short*)carve((size_t)HID * SIMID * 2);
    unsigned short* sWdT     = (unsigned short*)carve((size_t)SIMID * HID * 2);
    unsigned short* inter_r  = (unsigned short*)carve((size_t)T * 2 * IMID * 2);
    unsigned short* inter_sh = (unsigned short*)carve((size_t)T * SIMID * 2);

    const int mt = (T + 63) / 64;

    // ---- precision/layout prep: bf16 copies, weights transposed to [N][K] ----
    size_t nx4 = (size_t)T * HID / 4;
    cvt_kernel<<<(unsigned)((nx4 + 255) / 256), 256, 0, stream>>>(x, x_bf, nx4);
    transpose_cvt_kernel<<<dim3(IMID / 32, HID / 32, EXPERTS), 256, 0, stream>>>(Wg, WgT, HID, IMID);
    transpose_cvt_kernel<<<dim3(IMID / 32, HID / 32, EXPERTS), 256, 0, stream>>>(Wu, WuT, HID, IMID);
    transpose_cvt_kernel<<<dim3(HID / 32, IMID / 32, EXPERTS), 256, 0, stream>>>(Wd, WdT, IMID, HID);
    transpose_cvt_kernel<<<dim3(SIMID / 32, HID / 32, 1), 256, 0, stream>>>(sWg, sWgT, HID, SIMID);
    transpose_cvt_kernel<<<dim3(SIMID / 32, HID / 32, 1), 256, 0, stream>>>(sWu, sWuT, HID, SIMID);
    transpose_cvt_kernel<<<dim3(HID / 32, SIMID / 32, 1), 256, 0, stream>>>(sWd, sWdT, SIMID, HID);

    // ---- routing ----
    init_counts_kernel<<<1, 32, 0, stream>>>(counts);
    router_kernel<<<(T + 7) / 8, 256, 0, stream>>>(x, gate_w, T, topk_idx, topk_w, counts);
    scan_kernel<<<1, 32, 0, stream>>>(counts, offsets, cursors);
    scatter_kernel<<<(T + 255) / 256, 256, 0, stream>>>(topk_idx, topk_w, T, cursors,
                                                        pair_tok, pair_w);

    // ---- shared expert: writes every element of d_out (no zeroing needed) ----
    gateup_kernel<false><<<dim3(mt, SIMID / 64, 1), 256, 0, stream>>>(
        x_bf, sWgT, sWuT, nullptr, nullptr, nullptr, T, inter_sh);
    down_kernel<false><<<dim3(mt, HID / 64, 1), 256, 0, stream>>>(
        inter_sh, sWdT, nullptr, nullptr, nullptr, nullptr, T, out);

    // ---- routed experts: gathered rows per expert, atomic accumulate ----
    gateup_kernel<true><<<dim3(mt, IMID / 64, EXPERTS), 256, 0, stream>>>(
        x_bf, WgT, WuT, pair_tok, offsets, counts, 0, inter_r);
    down_kernel<true><<<dim3(mt, HID / 64, EXPERTS), 256, 0, stream>>>(
        inter_r, WdT, pair_tok, pair_w, offsets, counts, 0, out);
}
